// MultiHeadedAttention_31095563223751
// MI455X (gfx1250) — matrix operand order repaired and checked by reading, not compiled, not run on hardware
//
#include <hip/hip_runtime.h>
#include <hip/hip_bf16.h>

// Problem constants (match reference)
#define BSZ 4
#define LEN 512
#define DIM 1024
#define NH  16
#define DH  64

typedef __attribute__((ext_vector_type(16))) __bf16 bf16x16;
typedef __attribute__((ext_vector_type(8)))  float  f32x8;
typedef __attribute__((ext_vector_type(2)))  float  f32x2;
typedef __attribute__((ext_vector_type(2)))  __bf16 bf16x2;

union Frag { bf16x16 v; unsigned int u[8]; };
union Acc  { f32x8 v; float f[8]; };

#if __has_builtin(__builtin_amdgcn_global_load_async_to_lds_b128) && \
    __has_builtin(__builtin_amdgcn_s_wait_asynccnt)
#define ASYNC_OK 1
#else
#define ASYNC_OK 0
#endif

typedef __attribute__((__vector_size__(4 * sizeof(int)))) int v4i;
typedef __attribute__((address_space(1))) v4i* gv4i_p;   // global int4*
typedef __attribute__((address_space(3))) v4i* lv4i_p;   // LDS int4*

__device__ __forceinline__ void async_cp16(const void* gsrc, void* ldst) {
#if ASYNC_OK
  __builtin_amdgcn_global_load_async_to_lds_b128(
      (gv4i_p)const_cast<void*>(gsrc), (lv4i_p)ldst, 0, 0);
#else
  *(uint4*)ldst = *(const uint4*)gsrc;
#endif
}
__device__ __forceinline__ void wait_async() {
#if ASYNC_OK
  __builtin_amdgcn_s_wait_asynccnt(0);
#endif
}

// fp32x2 -> packed bf16 (one v_cvt_pk_bf16_f32)
__device__ __forceinline__ unsigned int pack2(float a, float b) {
#if __has_builtin(__builtin_amdgcn_cvt_pk_bf16_f32)
  union { bf16x2 v; unsigned int u; } t;
  t.v = __builtin_amdgcn_cvt_pk_bf16_f32(a, b);
  return t.u;
#else
  union { bf16x2 v; unsigned int u; } t;
  f32x2 s; s.x = a; s.y = b;
  t.v = __builtin_convertvector(s, bf16x2);
  return t.u;
#endif
}
__device__ __forceinline__ unsigned short f2bf(float f) {
  union { __bf16 h; unsigned short s; } t;
  t.h = (__bf16)f;
  return t.s;
}

// A fragment: 16x32 bf16 tile, rows m (0..15), K contiguous, row stride ld (ushorts).
__device__ __forceinline__ void load_a_frag(const unsigned short* As, int ld, int lane, Frag& a) {
  int m = lane & 15;
  int kh = (lane >> 4) << 3;
  const unsigned short* p0 = As + m * ld + kh;
  const unsigned short* p1 = As + m * ld + 16 + kh;
  uint4 x0 = *(const uint4*)p0;
  uint4 x1 = *(const uint4*)p1;
  a.u[0] = x0.x; a.u[1] = x0.y; a.u[2] = x0.z; a.u[3] = x0.w;
  a.u[4] = x1.x; a.u[5] = x1.y; a.u[6] = x1.z; a.u[7] = x1.w;
}
// B fragment: 32x16 bf16 tile column-major: Bs[n][k], k contiguous, col stride ld.
__device__ __forceinline__ void load_b_frag(const unsigned short* Bs, int ld, int lane, Frag& b) {
  int n = lane & 15;
  int k0 = (lane >> 4) << 3;
  const unsigned short* p = Bs + n * ld + k0;
  uint4 x0 = *(const uint4*)p;
  uint4 x1 = *(const uint4*)(p + 16);
  b.u[0] = x0.x; b.u[1] = x0.y; b.u[2] = x0.z; b.u[3] = x0.w;
  b.u[4] = x1.x; b.u[5] = x1.y; b.u[6] = x1.z; b.u[7] = x1.w;
}

__device__ __forceinline__ f32x8 wmma_bf16(const Frag& a, const Frag& b, f32x8 c) {
  return __builtin_amdgcn_wmma_f32_16x16x32_bf16(false, a.v, false, b.v, (short)0, c, false, false);
}

// ---------------------------------------------------------------------------
// fp32 -> bf16 table conversion (embeddings), n4 = count/4
// ---------------------------------------------------------------------------
__global__ __launch_bounds__(256)
void cvt_f32_bf16_kernel(const float* __restrict__ src, unsigned short* __restrict__ dst, int n4) {
  int i = blockIdx.x * 256 + threadIdx.x;
  if (i < n4) {
    const float4 v = ((const float4*)src)[i];
    uint2 t; t.x = pack2(v.x, v.y); t.y = pack2(v.z, v.w);
    ((uint2*)dst)[i] = t;
  }
}

// ---------------------------------------------------------------------------
// GEMM: Out = (X @ W + bias) * scale.  XBF selects bf16 input (async staged).
// out_mode: 0 = fp32 row-major MxN; 1 = bf16 [b][h][l][d]; 2 = bf16 [b][h][d][l].
// Block 256 = 8 waves; tile 128M x 64N; wave strip 16M x 64N (4 WMMAs/step).
// Double-buffered LDS: stage k+1 async while computing k; one barrier/step.
// ---------------------------------------------------------------------------
template<bool XBF>
__global__ __launch_bounds__(256)
void gemm_bf16_wmma(const float* __restrict__ Xf, const unsigned short* __restrict__ Xb,
                    const float* __restrict__ W, const float* __restrict__ bias,
                    void* __restrict__ Out, int M, int N, int K, float scale, int out_mode) {
  __shared__ alignas(16) unsigned short As[2][128 * 32];
  __shared__ alignas(16) unsigned short Bs[2][64 * 32];

  const int tid  = threadIdx.x;
  const int lane = tid & 31;
  const int wave = tid >> 5;
  const int m0   = blockIdx.x * 128;
  const int n0   = blockIdx.y * 64;

  auto stage = [&](int buf, int k0) {
    if (XBF) {
#pragma unroll
      for (int it = 0; it < 2; ++it) {
        int c = tid + it * 256;
        int row = c >> 2, c8 = (c & 3) << 3;
        async_cp16(Xb + (size_t)(m0 + row) * K + k0 + c8, As[buf] + row * 32 + c8);
      }
    } else {
#pragma unroll
      for (int it = 0; it < 4; ++it) {
        int idx = (tid + it * 256) << 2;
        int row = idx >> 5, col = idx & 31;
        const float4 a4 = *(const float4*)(Xf + (size_t)(m0 + row) * K + k0 + col);
        uint2 t; t.x = pack2(a4.x, a4.y); t.y = pack2(a4.z, a4.w);
        *(uint2*)(As[buf] + row * 32 + col) = t;
      }
    }
#pragma unroll
    for (int it = 0; it < 2; ++it) {
      int idx = (tid + it * 256) << 2;
      int k = idx >> 6, n = idx & 63;
      const float4 b4 = *(const float4*)(W + (size_t)(k0 + k) * N + n0 + n);
      Bs[buf][(n + 0) * 32 + k] = f2bf(b4.x);
      Bs[buf][(n + 1) * 32 + k] = f2bf(b4.y);
      Bs[buf][(n + 2) * 32 + k] = f2bf(b4.z);
      Bs[buf][(n + 3) * 32 + k] = f2bf(b4.w);
    }
  };

  Acc acc[4];
#pragma unroll
  for (int n = 0; n < 4; ++n)
#pragma unroll
    for (int r = 0; r < 8; ++r) acc[n].f[r] = 0.0f;

  stage(0, 0);
  for (int k0 = 0; k0 < K; k0 += 32) {
    const int cur = (k0 >> 5) & 1;
    wait_async();
    __syncthreads();                    // buffer 'cur' complete; prior reads done
    if (k0 + 32 < K) stage(cur ^ 1, k0 + 32);

    Frag a;
    load_a_frag(As[cur] + wave * 16 * 32, 32, lane, a);
#pragma unroll
    for (int n = 0; n < 4; ++n) {
      Frag b;
      load_b_frag(Bs[cur] + n * 16 * 32, 32, lane, b);
      acc[n].v = wmma_bf16(a, b, acc[n].v);
    }
  }

  const int cb = (lane & 15);
  const int rb = m0 + wave * 16 + ((lane >> 4) << 3);
#pragma unroll
  for (int n = 0; n < 4; ++n) {
    const int col = n0 + n * 16 + cb;
    const float bv = bias[col];
#pragma unroll
    for (int r = 0; r < 8; ++r) {
      int row = rb + r;
      float val = (acc[n].f[r] + bv) * scale;
      int b_ = row >> 9, l = row & (LEN - 1);
      int h  = col >> 6, d = col & (DH - 1);
      if (out_mode == 1) {
        ((unsigned short*)Out)[(((size_t)(b_ * NH + h) * LEN) + l) * DH + d] = f2bf(val);
      } else if (out_mode == 2) {
        ((unsigned short*)Out)[(((size_t)(b_ * NH + h) * DH) + d) * LEN + l] = f2bf(val);
      } else {
        ((float*)Out)[(size_t)row * N + col] = val;
      }
    }
  }
}

// ---------------------------------------------------------------------------
// qk: scores[b,h,i,j] = sum_d q[b,h,i,d]*k[b,h,j,d]; q,k bf16 (B,H,L,DH).
// grid (B*H, L/16, L/128); each wave one 16x16 j-tile, K=64. Async staging.
// ---------------------------------------------------------------------------
__global__ __launch_bounds__(256)
void qk_kernel(const unsigned short* __restrict__ qb, const unsigned short* __restrict__ kb,
               float* __restrict__ scores) {
  const int bh = blockIdx.x;
  const int i0 = blockIdx.y * 16;
  const int j0 = blockIdx.z * 128;
  __shared__ alignas(16) unsigned short Qs[16 * 64];   // [i][d]
  __shared__ alignas(16) unsigned short Ks[128 * 64];  // [j][d]

  const int tid = threadIdx.x, lane = tid & 31, wave = tid >> 5;
  if (tid < 128) {
    int row = tid >> 3, c8 = (tid & 7) << 3;
    async_cp16(qb + ((size_t)bh * LEN + i0 + row) * DH + c8, Qs + row * 64 + c8);
  }
#pragma unroll
  for (int it = 0; it < 4; ++it) {
    int c = tid + it * 256;
    int row = c >> 3, c8 = (c & 7) << 3;
    async_cp16(kb + ((size_t)bh * LEN + j0 + row) * DH + c8, Ks + row * 64 + c8);
  }
  wait_async();
  __syncthreads();

  Acc acc;
#pragma unroll
  for (int r = 0; r < 8; ++r) acc.f[r] = 0.0f;
#pragma unroll
  for (int kk = 0; kk < 64; kk += 32) {
    Frag a, b;
    load_a_frag(Qs + kk, 64, lane, a);
    load_b_frag(Ks + wave * 16 * 64 + kk, 64, lane, b);
    acc.v = wmma_bf16(a, b, acc.v);
  }
  const int j = j0 + wave * 16 + (lane & 15);
#pragma unroll
  for (int r = 0; r < 8; ++r) {
    int i = i0 + r + ((lane >> 4) << 3);
    scores[((size_t)bh * LEN + i) * LEN + j] = acc.f[r];
  }
}

// ---------------------------------------------------------------------------
// AST: scores[b,h,i,j] += sum_d q[b,h,i,d]*pe[b,i,j,d]; tables preconverted bf16.
// grid (B, L, L/128); A = 16 heads; each wave one 16-j tile. Async gather staging.
// ---------------------------------------------------------------------------
__global__ __launch_bounds__(256)
void ast_kernel(const unsigned short* __restrict__ qb, const int* __restrict__ ast,
                const unsigned short* __restrict__ seb, const unsigned short* __restrict__ veb,
                float* __restrict__ scores) {
  const int b  = blockIdx.x;
  const int i  = blockIdx.y;
  const int j0 = blockIdx.z * 128;
  __shared__ alignas(16) unsigned short Qs[16 * 64];   // [h][d]
  __shared__ alignas(16) unsigned short Ps[128 * 64];  // [j][d]

  const int tid = threadIdx.x, lane = tid & 31, wave = tid >> 5;
  if (tid < 128) {
    int h = tid >> 3, c8 = (tid & 7) << 3;
    async_cp16(qb + (((size_t)(b * NH + h)) * LEN + i) * DH + c8, Qs + h * 64 + c8);
  }
#pragma unroll
  for (int it = 0; it < 4; ++it) {
    int c = tid + it * 256;
    int jj = c >> 3, c8 = (c & 7) << 3;
    int j  = j0 + jj;
    int id = ast[((size_t)b * LEN + i) * LEN + j];
    const unsigned short* tab = ((j & 1) == 0) ? seb : veb;
    async_cp16(tab + (size_t)id * DH + c8, Ps + jj * 64 + c8);
  }
  wait_async();
  __syncthreads();

  Acc acc;
#pragma unroll
  for (int r = 0; r < 8; ++r) acc.f[r] = 0.0f;
#pragma unroll
  for (int kk = 0; kk < 64; kk += 32) {
    Frag a, bf;
    load_a_frag(Qs + kk, 64, lane, a);
    load_b_frag(Ps + wave * 16 * 64 + kk, 64, lane, bf);
    acc.v = wmma_bf16(a, bf, acc.v);
  }
  const int j = j0 + wave * 16 + (lane & 15);
#pragma unroll
  for (int r = 0; r < 8; ++r) {
    int h = r + ((lane >> 4) << 3);
    size_t addr = (((size_t)(b * NH + h)) * LEN + i) * LEN + j;
    scores[addr] += acc.f[r];
  }
}

// ---------------------------------------------------------------------------
// Softmax rows in place (fp32) + bf16 copy for the context GEMM.
// One wave per row; each lane owns 16 contiguous j.
// ---------------------------------------------------------------------------
__global__ __launch_bounds__(256)
void softmax_kernel(float* __restrict__ scores, unsigned short* __restrict__ scores_bf,
                    const unsigned char* __restrict__ mask) {
  const int lane = threadIdx.x & 31, wave = threadIdx.x >> 5;
  const size_t row = (size_t)blockIdx.x * 8 + wave;
  const int b = (int)(row >> 13);                         // row / (H*L)
  float* p = scores + row * LEN + lane * 16;
  unsigned short* pb = scores_bf + row * LEN + lane * 16;
  const unsigned char* mrow = mask + (size_t)b * LEN + lane * 16;

  float vals[16];
  const uint4 mw = *(const uint4*)mrow;
  float mx = -3.0e38f;
#pragma unroll
  for (int s = 0; s < 4; ++s) {
    const float4 v4 = *(const float4*)(p + 4 * s);
    vals[4 * s + 0] = v4.x; vals[4 * s + 1] = v4.y;
    vals[4 * s + 2] = v4.z; vals[4 * s + 3] = v4.w;
  }
  const unsigned int mb[4] = {mw.x, mw.y, mw.z, mw.w};
#pragma unroll
  for (int t = 0; t < 16; ++t) {
    if ((mb[t >> 2] >> ((t & 3) * 8)) & 0xFF) vals[t] = -1.0e18f;
    mx = fmaxf(mx, vals[t]);
  }
#pragma unroll
  for (int off = 16; off > 0; off >>= 1) mx = fmaxf(mx, __shfl_xor(mx, off, 32));
  float sum = 0.0f;
#pragma unroll
  for (int t = 0; t < 16; ++t) { vals[t] = __expf(vals[t] - mx); sum += vals[t]; }
#pragma unroll
  for (int off = 16; off > 0; off >>= 1) sum += __shfl_xor(sum, off, 32);
  const float inv = 1.0f / sum;
#pragma unroll
  for (int t = 0; t < 16; ++t) vals[t] *= inv;
#pragma unroll
  for (int s = 0; s < 4; ++s) {
    float4 v4; v4.x = vals[4 * s]; v4.y = vals[4 * s + 1];
    v4.z = vals[4 * s + 2]; v4.w = vals[4 * s + 3];
    *(float4*)(p + 4 * s) = v4;
  }
  uint4 o0, o1;
  o0.x = pack2(vals[0], vals[1]);  o0.y = pack2(vals[2], vals[3]);
  o0.z = pack2(vals[4], vals[5]);  o0.w = pack2(vals[6], vals[7]);
  o1.x = pack2(vals[8], vals[9]);  o1.y = pack2(vals[10], vals[11]);
  o1.z = pack2(vals[12], vals[13]); o1.w = pack2(vals[14], vals[15]);
  *(uint4*)pb = o0;
  *(uint4*)(pb + 8) = o1;
}

// ---------------------------------------------------------------------------
// context[b,l,h*64+d] (bf16) = sum_j attn_bf[b,h,l,j] * vt[b,h,d,j]
// grid (B*H, L/128); wave strip 16i x 64d; K over j step 32.
// Double-buffered pure-async staging; one barrier per step.
// ---------------------------------------------------------------------------
__global__ __launch_bounds__(256)
void context_kernel(const unsigned short* __restrict__ attn_bf,
                    const unsigned short* __restrict__ vtb,
                    unsigned short* __restrict__ ctxb) {
  const int bh = blockIdx.x;
  const int b  = bh >> 4, h = bh & 15;
  const int i0 = blockIdx.y * 128;
  __shared__ alignas(16) unsigned short As[2][128 * 32];  // [i][j]
  __shared__ alignas(16) unsigned short Vs[2][64 * 32];   // [d][j]

  const int tid = threadIdx.x, lane = tid & 31, wave = tid >> 5;

  auto stage = [&](int buf, int jk) {
#pragma unroll
    for (int it = 0; it < 2; ++it) {
      int c = tid + it * 256;
      int row = c >> 2, c8 = (c & 3) << 3;
      async_cp16(attn_bf + ((size_t)bh * LEN + i0 + row) * LEN + jk + c8,
                 As[buf] + row * 32 + c8);
    }
    int d = tid >> 2, c8 = (tid & 3) << 3;
    async_cp16(vtb + ((size_t)bh * DH + d) * LEN + jk + c8, Vs[buf] + d * 32 + c8);
  };

  Acc acc[4];
#pragma unroll
  for (int n = 0; n < 4; ++n)
#pragma unroll
    for (int r = 0; r < 8; ++r) acc[n].f[r] = 0.0f;

  stage(0, 0);
  for (int jk = 0; jk < LEN; jk += 32) {
    const int cur = (jk >> 5) & 1;
    wait_async();
    __syncthreads();
    if (jk + 32 < LEN) stage(cur ^ 1, jk + 32);

    Frag a;
    load_a_frag(As[cur] + wave * 16 * 32, 32, lane, a);
#pragma unroll
    for (int n = 0; n < 4; ++n) {
      Frag bf;
      load_b_frag(Vs[cur] + n * 16 * 32, 32, lane, bf);
      acc[n].v = wmma_bf16(a, bf, acc[n].v);
    }
  }

  const int rb = i0 + wave * 16 + ((lane >> 4) << 3);
#pragma unroll
  for (int n = 0; n < 4; ++n) {
    const int d = n * 16 + (lane & 15);
#pragma unroll
    for (int r = 0; r < 8; ++r) {
      int i = rb + r;
      ctxb[((size_t)b * LEN + i) * DIM + h * DH + d] = f2bf(acc[n].f[r]);
    }
  }
}

// ---------------------------------------------------------------------------
extern "C" void kernel_launch(void* const* d_in, const int* in_sizes, int n_in,
                              void* d_out, int out_size, void* d_ws, size_t ws_size,
                              hipStream_t stream) {
  const float* key   = (const float*)d_in[0];
  const float* value = (const float*)d_in[1];
  const float* query = (const float*)d_in[2];
  const int*   astm  = (const int*)d_in[3];
  const unsigned char* mask = (const unsigned char*)d_in[4];
  const float* Wq = (const float*)d_in[5];
  const float* bq = (const float*)d_in[6];
  const float* Wk = (const float*)d_in[7];
  const float* bk = (const float*)d_in[8];
  const float* Wv = (const float*)d_in[9];
  const float* bv = (const float*)d_in[10];
  const float* Wo = (const float*)d_in[11];
  const float* bo = (const float*)d_in[12];
  const float* struct_emb = (const float*)d_in[13];
  const float* value_emb  = (const float*)d_in[14];

  float* out_proj = (float*)d_out;                           // (B, L, D) fp32
  float* attn     = (float*)d_out + (size_t)BSZ * LEN * DIM; // (B, H, L, L) fp32

  // Workspace (bf16 intermediates)
  unsigned short* cur = (unsigned short*)d_ws;
  const size_t headsz = (size_t)BSZ * NH * LEN * DH;   // 2,097,152
  unsigned short* qb      = cur; cur += headsz;        // (B,H,L,DH)
  unsigned short* kb      = cur; cur += headsz;        // (B,H,L,DH)
  unsigned short* vtb     = cur; cur += headsz;        // (B,H,DH,L)
  unsigned short* ctxb    = cur; cur += headsz;        // (B,L,D)
  unsigned short* attn_bf = cur; cur += (size_t)BSZ * NH * LEN * LEN;
  unsigned short* seb     = cur; cur += 200 * DH;
  unsigned short* veb     = cur; cur += 10000 * DH;

  const int M = BSZ * LEN;      // 2048
  const float qscale = 0.125f;  // 1/sqrt(64)

  cvt_f32_bf16_kernel<<<(200 * DH / 4 + 255) / 256, 256, 0, stream>>>(struct_emb, seb, 200 * DH / 4);
  cvt_f32_bf16_kernel<<<(10000 * DH / 4 + 255) / 256, 256, 0, stream>>>(value_emb, veb, 10000 * DH / 4);

  dim3 gGemm(M / 128, DIM / 64);
  gemm_bf16_wmma<false><<<gGemm, 256, 0, stream>>>(query, nullptr, Wq, bq, qb,  M, DIM, DIM, qscale, 1);
  gemm_bf16_wmma<false><<<gGemm, 256, 0, stream>>>(key,   nullptr, Wk, bk, kb,  M, DIM, DIM, 1.0f,   1);
  gemm_bf16_wmma<false><<<gGemm, 256, 0, stream>>>(value, nullptr, Wv, bv, vtb, M, DIM, DIM, 1.0f,   2);

  dim3 gQK(BSZ * NH, LEN / 16, LEN / 128);
  qk_kernel<<<gQK, 256, 0, stream>>>(qb, kb, attn);

  dim3 gAST(BSZ, LEN, LEN / 128);
  ast_kernel<<<gAST, 256, 0, stream>>>(qb, astm, seb, veb, attn);

  softmax_kernel<<<(BSZ * NH * LEN) / 8, 256, 0, stream>>>(attn, attn_bf, mask);

  dim3 gCtx(BSZ * NH, LEN / 128);
  context_kernel<<<gCtx, 256, 0, stream>>>(attn_bf, vtb, ctxb);

  gemm_bf16_wmma<true><<<gGemm, 256, 0, stream>>>(nullptr, ctxb, Wo, bo, out_proj, M, DIM, DIM, 1.0f, 0);
}
